// RADT_77043123355755
// MI455X (gfx1250) — matrix operand order repaired and checked by reading, not compiled, not run on hardware
//
#include <hip/hip_runtime.h>
#include <hip/hip_bf16.h>
#include <cstdint>

#define BB 32
#define SS 128
#define DDIM 1024
#define HH 16
#define LL 6
#define VV 18
#define NN 256            // S * Q_LEN
#define HID 4096          // 4 * D
#define NEGV (-10000.0f)

typedef _Float16 h16;
typedef __attribute__((ext_vector_type(16))) _Float16 v16h;
typedef __attribute__((ext_vector_type(8)))  _Float16 v8h;
typedef __attribute__((ext_vector_type(4)))  _Float16 v4h;
typedef __attribute__((ext_vector_type(8)))  float    v8f;
typedef __attribute__((__vector_size__(16))) int      i4v;   // builtin's pointee type

// ---- CDNA5 async global->LDS copy (ASYNCcnt), with safe fallback ----------
#if defined(__has_builtin)
#  if __has_builtin(__builtin_amdgcn_global_load_async_to_lds_b128)
#    define USE_ASYNC_LDS 1
#  endif
#endif
#ifndef USE_ASYNC_LDS
#  define USE_ASYNC_LDS 0
#endif

__device__ __forceinline__ void wait_async0() {
#if defined(__has_builtin) && __has_builtin(__builtin_amdgcn_s_wait_asynccnt)
  __builtin_amdgcn_s_wait_asynccnt(0);
#else
  asm volatile("s_wait_asynccnt 0x0" ::: "memory");
#endif
}

// ---------------------------------------------------------------------------
// Generic WMMA GEMM:  out = act(A_f16[MxK] @ W_f32[KxNc] + bias) (+ resid)
// Block tile 128x128, 8 wave32 waves, each wave = 4x2 grid of 16x16 tiles.
// Double-buffered LDS: A tile streamed via GLOBAL_LOAD_ASYNC_TO_LDS_B128,
// W tile loaded as float4 (b128), 4x4 register transpose, ds_store_b64.
// GUARD=1 variant (scalar, bounds-checked W/stores) used only for the
// ragged head GEMM (Nc=18).
// ---------------------------------------------------------------------------
#define BM 128
#define BN 128
#define BKK 32
#define APITCH 40   // halves; 80B rows keep 16B alignment
#define BPITCH 40

template <int GUARD>
__global__ __launch_bounds__(256)
void gemm_wmma(const h16* __restrict__ A, const float* __restrict__ W,
               const float* __restrict__ bias, const float* __restrict__ resid,
               float* __restrict__ outF, h16* __restrict__ outH,
               int M, int K, int Nc, int act)
{
  __shared__ h16 As[2][BM * APITCH];
  __shared__ h16 Bs[2][BN * BPITCH];
  const int t  = threadIdx.x;
  const int m0 = blockIdx.y * BM;
  const int n0 = blockIdx.x * BN;
  const int wid  = t >> 5;
  const int lane = t & 31;
  const int g  = lane >> 4;        // half-wave group
  const int ml = lane & 15;
  const int wm = (wid & 1) * 64;   // wave M offset inside block
  const int wn = (wid >> 1) * 32;  // wave N offset inside block

  // W-staging assignment: thread covers 4 rows (k) x 4 cols (n)
  const int col4 = (t & 31) * 4;   // 0..124
  const int rowb = (t >> 5) * 4;   // 0..28

  v8f acc[4][2];
  #pragma unroll
  for (int i = 0; i < 4; i++)
    #pragma unroll
    for (int j = 0; j < 2; j++)
      #pragma unroll
      for (int r = 0; r < 8; r++) acc[i][j][r] = 0.0f;

  float wr[4][4];

  auto loadW = [&](int k0) {
    #pragma unroll
    for (int r = 0; r < 4; r++) {
      if constexpr (GUARD) {
        #pragma unroll
        for (int c = 0; c < 4; c++) {
          int n = n0 + col4 + c;
          wr[r][c] = (n < Nc) ? W[(size_t)(k0 + rowb + r) * Nc + n] : 0.0f;
        }
      } else {
        const float4 v = *(const float4*)(W + (size_t)(k0 + rowb + r) * Nc + n0 + col4);
        wr[r][0] = v.x; wr[r][1] = v.y; wr[r][2] = v.z; wr[r][3] = v.w;
      }
    }
  };
  auto storeW = [&](int buf) {           // 4x4 register transpose -> ds_store_b64
    #pragma unroll
    for (int c = 0; c < 4; c++) {
      v4h pk;
      pk[0] = (h16)wr[0][c]; pk[1] = (h16)wr[1][c];
      pk[2] = (h16)wr[2][c]; pk[3] = (h16)wr[3][c];
      *(v4h*)&Bs[buf][(col4 + c) * BPITCH + rowb] = pk;
    }
  };
  auto stageA = [&](int k0, int buf) {
    #pragma unroll
    for (int i = 0; i < 2; i++) {
      int chunk = t * 2 + i;             // 0..511, 8 halves each
      int row   = chunk >> 2;
      int c8    = (chunk & 3) * 8;
      h16* dst = &As[buf][row * APITCH + c8];
      const h16* src = A + (size_t)(m0 + row) * K + k0 + c8;
#if USE_ASYNC_LDS
      __builtin_amdgcn_global_load_async_to_lds_b128((i4v*)src, (i4v*)dst, 0, 0);
#else
      *(v8h*)dst = *(const v8h*)src;
#endif
    }
  };
  auto compute = [&](int buf) {
    v16h af[4], bf[2];
    #pragma unroll
    for (int i = 0; i < 4; i++) {
      int row = wm + i * 16 + ml;
      v8h lo = *(const v8h*)&As[buf][row * APITCH + 8 * g];
      v8h hi = *(const v8h*)&As[buf][row * APITCH + 16 + 8 * g];
      #pragma unroll
      for (int x = 0; x < 8; x++) { af[i][x] = lo[x]; af[i][x + 8] = hi[x]; }
    }
    #pragma unroll
    for (int j = 0; j < 2; j++) {
      int col = wn + j * 16 + ml;
      v8h lo = *(const v8h*)&Bs[buf][col * BPITCH + 8 * g];
      v8h hi = *(const v8h*)&Bs[buf][col * BPITCH + 16 + 8 * g];
      #pragma unroll
      for (int x = 0; x < 8; x++) { bf[j][x] = lo[x]; bf[j][x + 8] = hi[x]; }
    }
    #pragma unroll
    for (int i = 0; i < 4; i++)
      #pragma unroll
      for (int j = 0; j < 2; j++)
        acc[i][j] = __builtin_amdgcn_wmma_f32_16x16x32_f16(
            false, af[i], false, bf[j], (short)0, acc[i][j], false, false);
  };

  // ---- software-pipelined main loop (double-buffered LDS) ----
  loadW(0);
  stageA(0, 0);
  storeW(0);
#if USE_ASYNC_LDS
  wait_async0();
#endif
  __syncthreads();

  const int nsteps = K / BKK;
  for (int s = 0; s < nsteps; s++) {
    const int buf = s & 1, nxt = buf ^ 1;
    const bool more = (s + 1) < nsteps;
    if (more) {
      loadW((s + 1) * BKK);              // W f32 loads in flight during WMMAs
      stageA((s + 1) * BKK, nxt);        // async A tile -> LDS (ASYNCcnt)
    }
    if (!GUARD && (s + 2) < nsteps)      // keep the weight stream deep in L2
      __builtin_prefetch(W + (size_t)((s + 2) * BKK + rowb) * Nc + n0 + col4, 0, 1);
    compute(buf);
    if (more) storeW(nxt);
#if USE_ASYNC_LDS
    if (more) wait_async0();
#endif
    __syncthreads();
  }

  // ---- epilogue ----
  #pragma unroll
  for (int i = 0; i < 4; i++)
    #pragma unroll
    for (int j = 0; j < 2; j++) {
      int n = n0 + wn + j * 16 + ml;
      if (GUARD && n >= Nc) continue;
      #pragma unroll
      for (int r = 0; r < 8; r++) {
        int m = m0 + wm + i * 16 + 8 * g + r;
        float v = acc[i][j][r];
        if (bias)  v += bias[n];
        if (act == 1) v = 0.5f * v * (1.0f + erff(v * 0.70710678118f)); // exact GELU
        if (resid) v += resid[(size_t)m * Nc + n];
        if (outF)  outF[(size_t)m * Nc + n] = v;
        if (outH)  outH[(size_t)m * Nc + n] = (h16)v;
      }
    }
}

// ---------------------------------------------------------------------------
// Attention: one wave32 per (b, h, query-row). hd = 64.
// MK = key length (256 self / 128 cross), NDIVSH: mask m <= (n >> NDIVSH),
// PMSH: key-pad index = m >> PMSH.
// ---------------------------------------------------------------------------
template <int MK, int NDIVSH, int PMSH>
__global__ __launch_bounds__(256)
void attn_wave(const h16* __restrict__ qh, const h16* __restrict__ kh,
               const h16* __restrict__ vh, const int* __restrict__ pm,
               h16* __restrict__ outh)
{
  const int wid  = (blockIdx.x * 256 + threadIdx.x) >> 5;
  const int lane = threadIdx.x & 31;
  const int b  = wid / (HH * NN);
  const int r2 = wid % (HH * NN);
  const int h  = r2 / NN;
  const int n  = r2 % NN;

  const h16* qp = qh + ((size_t)(b * NN + n)) * DDIM + h * 64;
  float qv[64];
  #pragma unroll
  for (int d = 0; d < 64; d++) qv[d] = (float)qp[d];

  constexpr int CNT = MK >> 5;
  float p[CNT];
  float mx = -3.4e38f;
  const int nlim = n >> NDIVSH;
  #pragma unroll
  for (int jj = 0; jj < CNT; jj++) {
    int m = lane + (jj << 5);
    const h16* kp = kh + ((size_t)(b * MK + m)) * DDIM + h * 64;
    float s = 0.0f;
    #pragma unroll
    for (int d = 0; d < 64; d++) s += qv[d] * (float)kp[d];
    float sc = (m <= nlim) ? s * 0.125f : NEGV;       // hd^-0.5 = 1/8
    sc += (float)pm[b * SS + (m >> PMSH)] * NEGV;     // key padding bias
    p[jj] = sc;
    mx = fmaxf(mx, sc);
  }
  #pragma unroll
  for (int off = 16; off > 0; off >>= 1) mx = fmaxf(mx, __shfl_xor(mx, off, 32));
  float sum = 0.0f;
  #pragma unroll
  for (int jj = 0; jj < CNT; jj++) { p[jj] = expf(p[jj] - mx); sum += p[jj]; }
  #pragma unroll
  for (int off = 16; off > 0; off >>= 1) sum += __shfl_xor(sum, off, 32);
  const float inv = 1.0f / sum;

  float acc[64];
  #pragma unroll
  for (int d = 0; d < 64; d++) acc[d] = 0.0f;
  #pragma unroll
  for (int jj = 0; jj < CNT; jj++) {
    int m = lane + (jj << 5);
    const h16* vp = vh + ((size_t)(b * MK + m)) * DDIM + h * 64;
    float pj = p[jj];
    #pragma unroll
    for (int d = 0; d < 64; d++) acc[d] += pj * (float)vp[d];
  }
  h16* op = outh + ((size_t)(b * NN + n)) * DDIM + h * 64;
  #pragma unroll
  for (int d = 0; d < 64; d++) {
    float r = acc[d];
    #pragma unroll
    for (int off = 16; off > 0; off >>= 1) r += __shfl_xor(r, off, 32);
    if (lane == (d & 31)) op[d] = (h16)(r * inv);
  }
}

// ---------------------------------------------------------------------------
// AdaLN-modulated layernorm (eps 1e-6), in-place on f32 + f16 mirror.
// Row r of q uses mod row (r%N)/2 (repeat factor 2), chunks cS / cSh of 6D.
// ---------------------------------------------------------------------------
__global__ __launch_bounds__(256)
void ln_mod(float* __restrict__ x, h16* __restrict__ xh,
            const float* __restrict__ mod, int cS, int cSh)
{
  const int row  = blockIdx.x;       // [0, B*N)
  const int b    = row >> 8;         // N = 256
  const int n    = row & 255;
  const int srow = n >> 1;
  const float* sp  = mod + ((size_t)(b * SS + srow)) * (6 * DDIM) + (size_t)cS  * DDIM;
  const float* shp = mod + ((size_t)(b * SS + srow)) * (6 * DDIM) + (size_t)cSh * DDIM;
  float* xr = x  + (size_t)row * DDIM;
  h16*   xo = xh + (size_t)row * DDIM;
  const int t = threadIdx.x;

  __shared__ float r1[256], r2s[256];
  float vals[4];
  float s = 0.0f, ss = 0.0f;
  #pragma unroll
  for (int j = 0; j < 4; j++) {
    float v = xr[t + 256 * j];
    vals[j] = v; s += v; ss += v * v;
  }
  r1[t] = s; r2s[t] = ss; __syncthreads();
  for (int o = 128; o > 0; o >>= 1) {
    if (t < o) { r1[t] += r1[t + o]; r2s[t] += r2s[t + o]; }
    __syncthreads();
  }
  const float mean = r1[0] * (1.0f / DDIM);
  const float var  = r2s[0] * (1.0f / DDIM) - mean * mean;
  const float rstd = rsqrtf(var + 1e-6f);
  #pragma unroll
  for (int j = 0; j < 4; j++) {
    int d = t + 256 * j;
    float y = (vals[j] - mean) * rstd;
    y = y * (1.0f + sp[d]) + shp[d];
    xr[d] = y; xo[d] = (h16)y;
  }
}

// layernorm with gain/bias (eps 1e-5), optional +pe, f16 output only.
__global__ __launch_bounds__(256)
void ln_gb(const float* __restrict__ xin, const float* __restrict__ pe,
           const float* __restrict__ gg, const float* __restrict__ bb,
           h16* __restrict__ outh)
{
  const int row = blockIdx.x;        // [0, B*S)
  const float* xr = xin + (size_t)row * DDIM;
  const float* pr = pe ? pe + (size_t)row * DDIM : nullptr;
  h16* xo = outh + (size_t)row * DDIM;
  const int t = threadIdx.x;

  __shared__ float r1[256], r2s[256];
  float vals[4];
  float s = 0.0f, ss = 0.0f;
  #pragma unroll
  for (int j = 0; j < 4; j++) {
    int d = t + 256 * j;
    float v = xr[d] + (pr ? pr[d] : 0.0f);
    vals[j] = v; s += v; ss += v * v;
  }
  r1[t] = s; r2s[t] = ss; __syncthreads();
  for (int o = 128; o > 0; o >>= 1) {
    if (t < o) { r1[t] += r1[t + o]; r2s[t] += r2s[t + o]; }
    __syncthreads();
  }
  const float mean = r1[0] * (1.0f / DDIM);
  const float var  = r2s[0] * (1.0f / DDIM) - mean * mean;
  const float rstd = rsqrtf(var + 1e-5f);
  #pragma unroll
  for (int j = 0; j < 4; j++) {
    int d = t + 256 * j;
    float y = (vals[j] - mean) * rstd * gg[d] + bb[d];
    xo[d] = (h16)y;
  }
}

// ---------------------------- elementwise helpers --------------------------
__global__ void k_init_q(const float* __restrict__ q, float* __restrict__ qb,
                         h16* __restrict__ qhh)
{
  int i = blockIdx.x * 256 + threadIdx.x;
  float v = q[i];
  qb[i] = v; qhh[i] = (h16)v;
}
__global__ void k_silu(const float* __restrict__ c, h16* __restrict__ o)
{
  int i = blockIdx.x * 256 + threadIdx.x;
  float v = c[i];
  o[i] = (h16)(v / (1.0f + expf(-v)));
}
__global__ void k_combine(float* __restrict__ qb, const float* __restrict__ q2,
                          const float* __restrict__ al)
{
  int i = blockIdx.x * 256 + threadIdx.x;
  qb[i] = (1.0f + al[i]) * q2[i] + qb[i];
}

// ---------------------------------------------------------------------------
extern "C" void kernel_launch(void* const* d_in, const int* in_sizes, int n_in,
                              void* d_out, int out_size, void* d_ws, size_t ws_size,
                              hipStream_t stream)
{
  const float* in_q   = (const float*)d_in[0];
  const float* in_c   = (const float*)d_in[1];
  const float* in_pe  = (const float*)d_in[2];
  const float* mod_w  = (const float*)d_in[3];
  const float* mod_b  = (const float*)d_in[4];
  const float* sa_qw  = (const float*)d_in[5];
  const float* sa_kw  = (const float*)d_in[6];
  const float* sa_vw  = (const float*)d_in[7];
  const float* sa_pw  = (const float*)d_in[8];
  const float* sa_pb  = (const float*)d_in[9];
  const float* ca_qw  = (const float*)d_in[10];
  const float* ca_kw  = (const float*)d_in[11];
  const float* ca_vw  = (const float*)d_in[12];
  const float* ca_pw  = (const float*)d_in[13];
  const float* ca_pb  = (const float*)d_in[14];
  const float* nk_g   = (const float*)d_in[15];
  const float* nk_b   = (const float*)d_in[16];
  const float* nv_g   = (const float*)d_in[17];
  const float* nv_b   = (const float*)d_in[18];
  const float* al_w   = (const float*)d_in[19];
  const float* al_b   = (const float*)d_in[20];
  const float* fc1_w  = (const float*)d_in[21];
  const float* fc1_b  = (const float*)d_in[22];
  const float* fc2_w  = (const float*)d_in[23];
  const float* fc2_b  = (const float*)d_in[24];
  const float* head_w = (const float*)d_in[25];
  const float* head_b = (const float*)d_in[26];
  const int*   pmask  = (const int*)d_in[27];
  (void)in_sizes; (void)n_in; (void)out_size; (void)ws_size;

  size_t off = 0;
  auto take = [&](size_t bytes) -> char* {
    char* p = (char*)d_ws + off;
    off += (bytes + 255) & ~(size_t)255;
    return p;
  };
  const size_t BND = (size_t)BB * NN * DDIM;   // 8,388,608
  const size_t BSD = (size_t)BB * SS * DDIM;   // 4,194,304

  float* qbuf = (float*)take(BND * 4);                        // residual q (f32)
  float* q2b  = (float*)take(BND * 4);                        // cross-attn out q2
  float* alb  = (float*)take(BND * 4);                        // alpha
  float* modb = (float*)take((size_t)BB * SS * 6 * DDIM * 4); // AdaLN mod [B,S,6D]
  h16* q_h   = (h16*)take(BND * 2);
  h16* q2_h  = (h16*)take(BND * 2);
  h16* cs_h  = (h16*)take(BSD * 2);   // silu(c), constant across layers
  h16* kin_h = (h16*)take(BSD * 2);
  h16* vin_h = (h16*)take(BSD * 2);
  h16* qp_h  = (h16*)take(BND * 2);
  h16* kp_h  = (h16*)take(BND * 2);
  h16* vp_h  = (h16*)take(BND * 2);
  h16* att_h = (h16*)take(BND * 2);
  h16* hh    = (h16*)take((size_t)BB * NN * HID * 2);

  dim3 blk(256);
  k_init_q<<<(int)(BND / 256), blk, 0, stream>>>(in_q, qbuf, q_h);
  k_silu  <<<(int)(BSD / 256), blk, 0, stream>>>(in_c, cs_h);

  auto gemm = [&](const h16* A, const float* W, const float* bias,
                  const float* resid, float* oF, h16* oH,
                  int M, int K, int Nc, int act) {
    dim3 grid((Nc + BN - 1) / BN, M / BM);
    if (Nc % BN)
      gemm_wmma<1><<<grid, blk, 0, stream>>>(A, W, bias, resid, oF, oH, M, K, Nc, act);
    else
      gemm_wmma<0><<<grid, blk, 0, stream>>>(A, W, bias, resid, oF, oH, M, K, Nc, act);
  };

  const int Mq = BB * NN;   // 8192 rows (q stream)
  const int Mc = BB * SS;   // 4096 rows (c stream)
  const size_t D2 = (size_t)DDIM * DDIM;
  const int attnBlocks = (BB * HH * NN) / 8;  // 8 waves per 256-thread block

  for (int i = 0; i < LL; i++) {
    // ---- AdaLN modulation: mod = silu(c) @ mod_w + mod_b  [B,S,6D] ----
    gemm(cs_h, mod_w + (size_t)i * DDIM * 6 * DDIM, mod_b + (size_t)i * 6 * DDIM,
         nullptr, modb, nullptr, Mc, DDIM, 6 * DDIM, 0);

    // ---- block-causal self-attention + residual + modulated LN ----
    gemm(q_h, sa_qw + i * D2, nullptr, nullptr, nullptr, qp_h, Mq, DDIM, DDIM, 0);
    gemm(q_h, sa_kw + i * D2, nullptr, nullptr, nullptr, kp_h, Mq, DDIM, DDIM, 0);
    gemm(q_h, sa_vw + i * D2, nullptr, nullptr, nullptr, vp_h, Mq, DDIM, DDIM, 0);
    attn_wave<NN, 0, 1><<<attnBlocks, blk, 0, stream>>>(qp_h, kp_h, vp_h, pmask, att_h);
    gemm(att_h, sa_pw + i * D2, sa_pb + (size_t)i * DDIM, qbuf, qbuf, nullptr,
         Mq, DDIM, DDIM, 0);                                   // q = q1 + q
    ln_mod<<<Mq, blk, 0, stream>>>(qbuf, q_h, modb, 0, 1);     // s_sa, sh_sa

    // ---- cross attention to conditioning stream ----
    ln_gb<<<Mc, blk, 0, stream>>>(in_c, in_pe, nk_g + (size_t)i * DDIM,
                                  nk_b + (size_t)i * DDIM, kin_h);
    ln_gb<<<Mc, blk, 0, stream>>>(in_c, nullptr, nv_g + (size_t)i * DDIM,
                                  nv_b + (size_t)i * DDIM, vin_h);
    gemm(q_h,   ca_qw + i * D2, nullptr, nullptr, nullptr, qp_h, Mq, DDIM, DDIM, 0);
    gemm(kin_h, ca_kw + i * D2, nullptr, nullptr, nullptr, kp_h, Mc, DDIM, DDIM, 0);
    gemm(vin_h, ca_vw + i * D2, nullptr, nullptr, nullptr, vp_h, Mc, DDIM, DDIM, 0);
    attn_wave<SS, 1, 0><<<attnBlocks, blk, 0, stream>>>(qp_h, kp_h, vp_h, pmask, att_h);
    gemm(att_h, ca_pw + i * D2, ca_pb + (size_t)i * DDIM, nullptr, q2b, q2_h,
         Mq, DDIM, DDIM, 0);
    // alpha = concat(q2, q) @ al_w + al_b  (two accumulating GEMMs)
    gemm(q2_h, al_w + (size_t)i * 2 * D2,      al_b + (size_t)i * DDIM, nullptr,
         alb, nullptr, Mq, DDIM, DDIM, 0);
    gemm(q_h,  al_w + (size_t)i * 2 * D2 + D2, nullptr, alb,
         alb, nullptr, Mq, DDIM, DDIM, 0);
    k_combine<<<(int)(BND / 256), blk, 0, stream>>>(qbuf, q2b, alb); // q=(1+a)*q2+q
    ln_mod<<<Mq, blk, 0, stream>>>(qbuf, q_h, modb, 2, 3);     // s_ca, sh_ca

    // ---- MLP: gelu(q@fc1+b1)@fc2 + b2 + residual, then modulated LN ----
    gemm(q_h, fc1_w + (size_t)i * DDIM * HID, fc1_b + (size_t)i * HID, nullptr,
         nullptr, hh, Mq, DDIM, HID, 1);                       // exact GELU
    gemm(hh,  fc2_w + (size_t)i * HID * DDIM, fc2_b + (size_t)i * DDIM, qbuf,
         qbuf, nullptr, Mq, HID, DDIM, 0);
    ln_mod<<<Mq, blk, 0, stream>>>(qbuf, q_h, modb, 4, 5);     // s_mlp, sh_mlp
  }

  // ---- head: logits = q @ head_w + head_b  [B,N,18] ----
  gemm(q_h, head_w, head_b, nullptr, (float*)d_out, nullptr, Mq, DDIM, VV, 0);
}
